// HybridBandAttn_79645873537310
// MI455X (gfx1250) — matrix operand order repaired and checked by reading, not compiled, not run on hardware
//
#include <hip/hip_runtime.h>
#include <hip/hip_bf16.h>

// ---------------------------------------------------------------------------
// HybridBandAttn for MI455X (gfx1250, wave32, WMMA, async-to-LDS).
//   x[N,4,128] --async DMA--> LDS -> qkv (bf16 WMMA, transposed LDS out)
//   -> 4x4 per-head softmax attn (packed b64 LDS reads) -> proj (bf16 WMMA)
//   -> residual (fp32) -> [B,512,H,W] channel-major coalesced stores.
// Memory-bound (~268MB @ 23.3TB/s => ~11.5us floor): global I/O fp32
// coalesced; matmuls on v_wmma_f32_16x16x32_bf16 with f32 accumulation.
// ---------------------------------------------------------------------------

#define HW        16384     // H*W = 128*128
#define TILE      16        // pixels per workgroup
#define ROWS      64        // token rows per workgroup (TILE*4)
#define PX        132       // xf fp32 pitch (floats); stride%64=4 -> A-frag reads conflict-free
#define PXY       132       // yst fp32 pitch
#define QP2       72        // qkvT pitch in tokens (bf16); 36-dword stride -> conflict-free col stores
#define OP        68        // ostT pitch in tokens (f32); 68-dword stride -> conflict-free col stores
#define WSQ_FRAGS (24*4)    // w_qkv fragments: 24 col-tiles x 4 k-tiles
#define WSQ_ELEMS (WSQ_FRAGS*512)
#define SCALEF    0.17677669529663687f  // HEAD_DIM^-0.5, HEAD_DIM=32

typedef __attribute__((ext_vector_type(16))) __bf16 v16bf;
typedef __attribute__((ext_vector_type(8)))  __bf16 v8bf;
typedef __attribute__((ext_vector_type(4)))  __bf16 v4bf;
typedef __attribute__((ext_vector_type(8)))  float  v8f;
typedef __attribute__((ext_vector_type(4)))  float  v4f;

// Build an A-fragment (16x32 bf16, M=lane&15) from an fp32 LDS row.
// ISA layout: element e of lane l holds K = e + (e>=8?8:0) + (l>=16?8:0).
__device__ __forceinline__ v16bf a_frag_f32(const float* __restrict__ row,
                                            int c0, int lhi8) {
  v16bf a;
  const float* p1 = row + c0 + lhi8;        // K = c0 + lhi8 + [0..7]
  const float* p2 = p1 + 16;                // K = c0 + 16 + lhi8 + [0..7]
#pragma unroll
  for (int e = 0; e < 8; ++e) {
    a[e]     = (__bf16)p1[e];
    a[e + 8] = (__bf16)p2[e];
  }
  return a;
}

// ---------------------------------------------------------------------------
// Pre-swizzle weights into bf16 WMMA B-fragment layout in d_ws.
// B (32x16, KxN): lane l holds N = l&15, K = (e&7) + (e>=8?16:0) + (l>=16?8:0), e=0..15.
// Fragment f=(ct*4+kt) at ws[f*512 + lane*16 + e] -> one contiguous 32B
// chunk per lane (fully coalesced, L2-resident: 128KB << 192MB L2).
// ---------------------------------------------------------------------------
__global__ __launch_bounds__(256) void hba_prep_weights(
    const float* __restrict__ wq,   // [384,128]
    const float* __restrict__ wp,   // [128,128]
    __bf16* __restrict__ ws) {
  const int idx  = blockIdx.x * 256 + threadIdx.x;   // 0 .. 65535
  const int lane = (idx >> 4) & 31;
  const int e    = idx & 15;
  const int nlo  = lane & 15;
  const int kk   = (e & 7) + ((e >= 8) ? 16 : 0) + ((lane >= 16) ? 8 : 0);
  if (idx < WSQ_ELEMS) {
    const int frag = idx >> 9;                       // ct*4 + kt
    const int ct = frag >> 2, kt = frag & 3;
    ws[idx] = (__bf16)wq[(ct * 16 + nlo) * 128 + kt * 32 + kk];
  } else {
    const int j    = idx - WSQ_ELEMS;                // w_proj fragments
    const int frag = j >> 9;
    const int ct = frag >> 2, kt = frag & 3;
    ws[idx] = (__bf16)wp[(ct * 16 + nlo) * 128 + kt * 32 + kk];
  }
}

// ---------------------------------------------------------------------------
// Main kernel: 128 threads (4 wave32), 16 pixels / workgroup.
// ---------------------------------------------------------------------------
__global__ __launch_bounds__(128) void hba_main(
    const float*  __restrict__ ll,      // [4,128,128,128]
    const float*  __restrict__ hf,      // [4,384,128,128]
    const float*  __restrict__ b_proj,  // [128]
    const float*  __restrict__ rs_ptr,  // scalar
    const __bf16* __restrict__ wsfrag,  // pre-swizzled weights
    float*        __restrict__ out) {   // [4,512,128,128]

  __shared__ __align__(16) float  xf[ROWS * PX];      // fp32 x tile (residual)
  __shared__ __align__(16) __bf16 qkvT[384 * QP2];    // qkv, [d][token];
                                                      // reused as ostT fp32
  __shared__ __align__(16) float  yst[ROWS * PXY];    // attn output (fp32)

  const int tid  = threadIdx.x;
  const int lane = tid & 31;
  const int wv   = tid >> 5;                 // wave id 0..3
  const int n0   = blockIdx.x * TILE;        // first pixel of tile
  const int b    = n0 >> 14;                 // batch (HW = 16384)
  const int pixbase = n0 & (HW - 1);

  // ---- Phase 0: DMA x tile into LDS with the transpose folded into the
  //      per-lane LDS destination (global_load_async_to_lds_b32). ----
  {
    const int p = tid & 15;                  // pixel in tile
    const int g = tid >> 4;                  // channel group 0..7
#pragma unroll 4
    for (int it = 0; it < 64; ++it) {
      const int sc = it * 8 + g;             // 0..511 = (s,c)
      const int s = sc >> 7;
      const int c = sc & 127;
      const float* src = (s == 0)
          ? (ll + ((size_t)b * 128 + c) * HW + pixbase)
          : (hf + ((size_t)b * 384 + (size_t)(s - 1) * 128 + c) * HW + pixbase);
      const unsigned long long ga = (unsigned long long)(src + p);
      // Low 32 bits of a generic shared pointer == LDS byte offset.
      const unsigned ldsoff = (unsigned)(size_t)&xf[(p * 4 + s) * PX + c];
      asm volatile("global_load_async_to_lds_b32 %0, %1, off"
                   :: "v"(ldsoff), "v"(ga) : "memory");
    }
    asm volatile("s_wait_asynccnt 0" ::: "memory");
  }
  __syncthreads();

  const int nlo  = lane & 15;
  const int lhi8 = (lane >= 16) ? 8 : 0;     // K/M offset for upper half-wave
  const int rowbase = wv * 16;

  // ---- Phase 1: QKV GEMM (16 rows x 384) per wave, bf16 WMMA.
  //      D stored transposed: 8 accum elems = 8 consecutive tokens ->
  //      one packed-bf16 ds_store_b128 per column. ----
  {
    const float* arow = xf + (rowbase + nlo) * PX;
    v16bf a[4];
#pragma unroll
    for (int kt = 0; kt < 4; ++kt) a[kt] = a_frag_f32(arow, kt * 32, lhi8);

    const v16bf* bw = (const v16bf*)wsfrag;  // fragment f at bw[f*32 + lane]
#pragma unroll 2
    for (int ct = 0; ct < 24; ++ct) {
      v8f acc = {0.f, 0.f, 0.f, 0.f, 0.f, 0.f, 0.f, 0.f};
#pragma unroll
      for (int kt = 0; kt < 4; ++kt) {
        v16bf bf = bw[(ct * 4 + kt) * 32 + lane];
        acc = __builtin_amdgcn_wmma_f32_16x16x32_bf16(
            false, a[kt], false, bf, (short)0, acc, false, false);
      }
      v8bf pk;
#pragma unroll
      for (int r = 0; r < 8; ++r) pk[r] = (__bf16)acc[r];
      *(v8bf*)&qkvT[(ct * 16 + nlo) * QP2 + rowbase + lhi8] = pk;
    }
  }
  __syncthreads();

  // ---- Phase 2: per-(pixel,head) 4x4 softmax attention (VALU).
  //      qkvT gives all 4 tokens of one head-dim in a single b64 load. ----
  if (tid < 64) {
    const int p  = tid & 15;                 // consecutive lanes -> 8B runs
    const int hd = tid >> 4;
    const int r0 = p * 4;
    const int d0 = hd * 32;
    float sc[4][4];
#pragma unroll
    for (int i = 0; i < 4; ++i)
#pragma unroll
      for (int j = 0; j < 4; ++j) sc[i][j] = 0.f;

    for (int d = 0; d < 32; ++d) {
      const v4bf q4 = *(const v4bf*)&qkvT[(d0 + d) * QP2 + r0];
      const v4bf k4 = *(const v4bf*)&qkvT[(128 + d0 + d) * QP2 + r0];
      float qv[4], kv[4];
#pragma unroll
      for (int s = 0; s < 4; ++s) { qv[s] = (float)q4[s]; kv[s] = (float)k4[s]; }
#pragma unroll
      for (int i = 0; i < 4; ++i)
#pragma unroll
        for (int j = 0; j < 4; ++j) sc[i][j] += qv[i] * kv[j];
    }
    float wgt[4][4];
#pragma unroll
    for (int i = 0; i < 4; ++i) {
      float m = -1e30f;
#pragma unroll
      for (int j = 0; j < 4; ++j) { sc[i][j] *= SCALEF; m = fmaxf(m, sc[i][j]); }
      float sum = 0.f;
#pragma unroll
      for (int j = 0; j < 4; ++j) { wgt[i][j] = __expf(sc[i][j] - m); sum += wgt[i][j]; }
      const float inv = 1.f / sum;
#pragma unroll
      for (int j = 0; j < 4; ++j) wgt[i][j] *= inv;
    }
    // y = attn @ v, 4 head-dims at a time -> float4 stores into yst.
#pragma unroll 2
    for (int dblk = 0; dblk < 8; ++dblk) {
      float vv[4][4];                        // [dd][token]
#pragma unroll
      for (int dd = 0; dd < 4; ++dd) {
        const v4bf v4 = *(const v4bf*)&qkvT[(256 + d0 + dblk * 4 + dd) * QP2 + r0];
#pragma unroll
        for (int s = 0; s < 4; ++s) vv[dd][s] = (float)v4[s];
      }
#pragma unroll
      for (int i = 0; i < 4; ++i) {
        v4f y;
#pragma unroll
        for (int dd = 0; dd < 4; ++dd)
          y[dd] = wgt[i][0] * vv[dd][0] + wgt[i][1] * vv[dd][1] +
                  wgt[i][2] * vv[dd][2] + wgt[i][3] * vv[dd][3];
        *(v4f*)&yst[(r0 + i) * PXY + d0 + dblk * 4] = y;
      }
    }
  }
  __syncthreads();

  // ---- Phase 3: projection GEMM (16 rows x 128), bf16 WMMA.
  //      D stored transposed fp32: whole v8f accumulator = 32B contiguous. ----
  {
    const float* arow = yst + (rowbase + nlo) * PXY;
    v16bf a[4];
#pragma unroll
    for (int kt = 0; kt < 4; ++kt) a[kt] = a_frag_f32(arow, kt * 32, lhi8);

    const v16bf* bw = ((const v16bf*)wsfrag) + WSQ_FRAGS * 32;  // w_proj frags
    float* ostT = (float*)qkvT;              // reuse qkv LDS region
#pragma unroll 2
    for (int ct = 0; ct < 8; ++ct) {
      v8f acc = {0.f, 0.f, 0.f, 0.f, 0.f, 0.f, 0.f, 0.f};
#pragma unroll
      for (int kt = 0; kt < 4; ++kt) {
        v16bf bf = bw[(ct * 4 + kt) * 32 + lane];
        acc = __builtin_amdgcn_wmma_f32_16x16x32_bf16(
            false, a[kt], false, bf, (short)0, acc, false, false);
      }
      *(v8f*)&ostT[(ct * 16 + nlo) * OP + rowbase + lhi8] = acc;
    }
  }
  __syncthreads();

  // ---- Phase 4: bias + residual, channel-major coalesced store. ----
  {
    const float rsv = rs_ptr[0];
    const float* ostT = (const float*)qkvT;
    const int p = tid & 15;
    const int g = tid >> 4;
#pragma unroll 4
    for (int it = 0; it < 64; ++it) {
      const int sc = it * 8 + g;             // 0..511 = (s,d)
      const int s = sc >> 7;
      const int d = sc & 127;
      const int row = p * 4 + s;
      const float xv = xf[row * PX + d];
      const float yv = ostT[d * OP + row] + b_proj[d];
      out[((size_t)b * 512 + (size_t)(s * 128 + d)) * HW + pixbase + p] =
          xv + rsv * yv;
    }
  }
}

extern "C" void kernel_launch(void* const* d_in, const int* in_sizes, int n_in,
                              void* d_out, int out_size, void* d_ws, size_t ws_size,
                              hipStream_t stream) {
  const float* ll = (const float*)d_in[0];
  const float* hf = (const float*)d_in[1];
  const float* wq = (const float*)d_in[2];
  const float* wp = (const float*)d_in[3];
  const float* bp = (const float*)d_in[4];
  const float* rs = (const float*)d_in[5];
  __bf16* ws = (__bf16*)d_ws;                 // needs 128KB of scratch
  float* out = (float*)d_out;

  // Pre-swizzle both weight matrices into WMMA B-fragment layout (bf16).
  hba_prep_weights<<<256, 256, 0, stream>>>(wq, wp, ws);

  const int npix   = in_sizes[0] / 128;       // B*H*W
  const int blocks = npix / TILE;
  hba_main<<<blocks, 128, 0, stream>>>(ll, hf, bp, rs, ws, out);
}
